// convlstm_91139206021172
// MI455X (gfx1250) — matrix operand order repaired and checked by reading, not compiled, not run on hardware
//
#include <hip/hip_runtime.h>
#include <hip/hip_bf16.h>
#include <stdint.h>

// ---------------- problem constants ----------------
#define BB 128   // batch
#define TT 128   // time steps
#define LL 128   // spatial length
#define HH 32    // hidden channels
#define CC 64    // combined channels (x:32 | h:32)
#define NO 128   // 4*H gate outputs
#define KC 192   // 3 taps * CC  (GEMM K dim, = 6 WMMA k-steps)
#define LPAD (LL + 2)

#define SMEM_WT_BYTES   (NO * KC * 2)       // 49152
#define SMEM_COMB_BYTES (LPAD * CC * 2)     // 16640
#define SMEM_TOTAL      (SMEM_WT_BYTES + SMEM_COMB_BYTES)  // 65792 (< 320KB WGP LDS)

typedef __bf16 bf16_t;
typedef __attribute__((ext_vector_type(16))) __bf16 v16bf;
typedef __attribute__((ext_vector_type(8)))  float  v8f;
typedef __attribute__((ext_vector_type(8)))  unsigned int v8u;

__device__ inline bf16_t f2bf(float f) { return (bf16_t)f; }
__device__ inline float sigm(float x) { return 1.0f / (1.0f + __expf(-x)); }

// ---- A fragment (16x32 bf16), contiguous K layout (matches B):
// lane<16 : elems0..15 = K kb+0..15
// lane>=16: elems0..15 = K kb+16..31
__device__ inline v16bf load_frag_a_bf16(const bf16_t* row, int kbase, int khalf) {
  const uint4* p0 = reinterpret_cast<const uint4*>(row + kbase + khalf * 16);
  const uint4* p1 = reinterpret_cast<const uint4*>(row + kbase + khalf * 16 + 8);
  uint4 q0 = *p0, q1 = *p1;
  v8u t;
  t[0] = q0.x; t[1] = q0.y; t[2] = q0.z; t[3] = q0.w;
  t[4] = q1.x; t[5] = q1.y; t[6] = q1.z; t[7] = q1.w;
  return __builtin_bit_cast(v16bf, t);
}

// ---- B fragment (32x16 bf16): lane holds col = lane%16,
// K rows kb + (lane>=16 ? 16 : 0) + 0..15 : 16 consecutive bf16 (32B).
__device__ inline v16bf load_frag_b_bf16(const bf16_t* p) {
  const uint4* q = reinterpret_cast<const uint4*>(p);
  uint4 q0 = q[0], q1 = q[1];
  v8u t;
  t[0] = q0.x; t[1] = q0.y; t[2] = q0.z; t[3] = q0.w;
  t[4] = q1.x; t[5] = q1.y; t[6] = q1.z; t[7] = q1.w;
  return __builtin_bit_cast(v16bf, t);
}

// =====================================================================
// Whole ConvLSTM scan in ONE launch. Block = one batch element, 8 waves.
// Per step: GEMM M=128(L) N=128(4H) K=192 from LDS (48 v_wmma per wave),
// gates fully in registers (D-fragment holds all 4 gate channels per lane),
// h_{t+1} + fused input-conv x_{t+1} written back to LDS comb.
// Cell state c never leaves registers until the final store (in reshape order).
// =====================================================================
__global__ __launch_bounds__(256) void lstm_scan(
    float* __restrict__ cflat,
    const float* __restrict__ x, const float* __restrict__ w_in,
    const float* __restrict__ b_in, const float* __restrict__ w_cell) {
  extern __shared__ char smem[];
  bf16_t* s_wT   = reinterpret_cast<bf16_t*>(smem);                  // [NO][KC]
  bf16_t* s_comb = reinterpret_cast<bf16_t*>(smem + SMEM_WT_BYTES);  // [LPAD][CC]

  const int b     = blockIdx.x;
  const int tid   = threadIdx.x;
  const int mt    = tid >> 5;       // wave id = M tile (8 x 16 rows = L)
  const int lane  = tid & 31;
  const int khalf = lane >> 4;
  const int lr    = lane & 15;

  // ---- stage cell weights into LDS once: s_wT[o][k*64+i] = w_cell[o][i][k]
  for (int e = tid; e < NO * KC; e += 256) {
    int o = e / KC, r = e % KC;
    int k = r / CC, ci = r % CC;
    s_wT[e] = f2bf(w_cell[((size_t)o * CC + ci) * 3 + k]);
  }
  // ---- zero comb (including the two zero-pad rows)
  for (int e = tid; e < LPAD * CC; e += 256) s_comb[e] = f2bf(0.0f);

  // per-lane input-conv weights for its two hidden channels
  const int h0 = lr, h1 = lr + 16;
  float wi0[3], wi1[3];
  #pragma unroll
  for (int k = 0; k < 3; ++k) { wi0[k] = w_in[h0 * 3 + k]; wi1[k] = w_in[h1 * 3 + k]; }
  const float bi0 = b_in[h0], bi1 = b_in[h1];

  const int l0 = mt * 16 + (khalf << 3);   // first of this lane's 8 gate rows
  __syncthreads();

  // ---- x_0 section of comb (fused input conv at t=0)
  {
    const float* xrow = x + ((size_t)b * TT) * LL;
    float xb[10];
    #pragma unroll
    for (int j = 0; j < 10; ++j) {
      int li = l0 - 1 + j;
      xb[j] = (li >= 0 && li < LL) ? xrow[li] : 0.0f;
    }
    #pragma unroll
    for (int r = 0; r < 8; ++r) {
      float xv0 = bi0, xv1 = bi1;
      #pragma unroll
      for (int k = 0; k < 3; ++k) { xv0 += wi0[k] * xb[r + k]; xv1 += wi1[k] * xb[r + k]; }
      s_comb[(l0 + r + 1) * CC + h0] = f2bf(xv0);
      s_comb[(l0 + r + 1) * CC + h1] = f2bf(xv1);
    }
  }

  float c0[8], c1[8];
  #pragma unroll
  for (int r = 0; r < 8; ++r) { c0[r] = 0.0f; c1[r] = 0.0f; }

  // A row for this lane: row m = lane%16 -> position l = mt*16+lr;
  // padded comb row l spans positions l-1..l+1 contiguously (192 bf16).
  const bf16_t* arow = s_comb + (size_t)(mt * 16 + lr) * CC;

  __syncthreads();

  for (int t = 0; t < TT; ++t) {
    // -------- GEMM: cc(L x 4H) = comb_patches(L x 192) @ wT^T --------
    v8f acc[8];
    #pragma unroll
    for (int i = 0; i < 8; ++i) acc[i] = (v8f){0, 0, 0, 0, 0, 0, 0, 0};

    #pragma unroll
    for (int kk = 0; kk < 6; ++kk) {
      v16bf af = load_frag_a_bf16(arow, kk * 32, khalf);
      #pragma unroll
      for (int nt = 0; nt < 8; ++nt) {
        const bf16_t* bp = s_wT + (size_t)(nt * 16 + lr) * KC + kk * 32 + khalf * 16;
        v16bf bfrag = load_frag_b_bf16(bp);
        acc[nt] = __builtin_amdgcn_wmma_f32_16x16x32_bf16(
            false, af, false, bfrag, (short)0, acc[nt], false, false);
      }
    }
    __syncthreads();   // all waves done reading s_comb for this step

    // -------- gates in registers + fused x_{t+1} conv --------
    // D layout: acc[nt][r] -> row m = r + khalf*8 (l = l0+r), col n = nt*16+lr.
    // Gate channels for h0=lr: nt 0(i),2(f),4(o),6(g); for h1=lr+16: nt 1,3,5,7.
    const int tn = t + 1;
    float xb[10];
    if (tn < TT) {
      const float* xrow = x + ((size_t)b * TT + tn) * LL;
      #pragma unroll
      for (int j = 0; j < 10; ++j) {
        int li = l0 - 1 + j;
        xb[j] = (li >= 0 && li < LL) ? xrow[li] : 0.0f;
      }
    }
    #pragma unroll
    for (int r = 0; r < 8; ++r) {
      const int lp = l0 + r + 1;  // padded comb row for position l0+r
      {
        float ci = acc[0][r], cf = acc[2][r], co = acc[4][r], cg = acc[6][r];
        float cn = sigm(cf) * c0[r] + sigm(ci) * tanhf(cg);
        c0[r] = cn;
        s_comb[lp * CC + HH + h0] = f2bf(sigm(co) * tanhf(cn));
      }
      {
        float ci = acc[1][r], cf = acc[3][r], co = acc[5][r], cg = acc[7][r];
        float cn = sigm(cf) * c1[r] + sigm(ci) * tanhf(cg);
        c1[r] = cn;
        s_comb[lp * CC + HH + h1] = f2bf(sigm(co) * tanhf(cn));
      }
      if (tn < TT) {
        float xv0 = bi0, xv1 = bi1;
        #pragma unroll
        for (int k = 0; k < 3; ++k) { xv0 += wi0[k] * xb[r + k]; xv1 += wi1[k] * xb[r + k]; }
        s_comb[lp * CC + h0] = f2bf(xv0);
        s_comb[lp * CC + h1] = f2bf(xv1);
      }
    }
    __syncthreads();   // comb ready for next step's GEMM
  }

  // ---- emit c_T directly in reshape(B, H*L) order (h-major)
  #pragma unroll
  for (int r = 0; r < 8; ++r) {
    int l = l0 + r;
    cflat[(size_t)b * (HH * LL) + h0 * LL + l] = c0[r];
    cflat[(size_t)b * (HH * LL) + h1 * LL + l] = c1[r];
  }
}

// =====================================================================
// MLP GEMM: out(MxN) = A(MxK) @ W(NxK)^T + bias ; f32 in/out, bf16 WMMA.
// One wave computes a 16x64 tile (A fragment reused across 4 N tiles).
// =====================================================================
__global__ __launch_bounds__(32) void mlp_gemm(
    const float* __restrict__ A, const float* __restrict__ W,
    const float* __restrict__ bias, float* __restrict__ out,
    int M, int N, int K) {
  int nq    = N >> 6;
  int mt    = blockIdx.x / nq;
  int ntq   = blockIdx.x % nq;
  int lane  = threadIdx.x;
  int khalf = lane >> 4;
  int lr    = lane & 15;

  v8f acc[4];
  #pragma unroll
  for (int j = 0; j < 4; ++j) acc[j] = (v8f){0, 0, 0, 0, 0, 0, 0, 0};

  const float* arow = A + (size_t)(mt * 16 + lr) * K;

  for (int kb = 0; kb < K; kb += 32) {
    v16bf af;
    {
      const float4* p0 = reinterpret_cast<const float4*>(arow + kb + khalf * 16);
      const float4* p1 = reinterpret_cast<const float4*>(arow + kb + khalf * 16 + 8);
      float4 q0 = p0[0], q1 = p0[1], q2 = p1[0], q3 = p1[1];
      af[0]  = f2bf(q0.x); af[1]  = f2bf(q0.y); af[2]  = f2bf(q0.z); af[3]  = f2bf(q0.w);
      af[4]  = f2bf(q1.x); af[5]  = f2bf(q1.y); af[6]  = f2bf(q1.z); af[7]  = f2bf(q1.w);
      af[8]  = f2bf(q2.x); af[9]  = f2bf(q2.y); af[10] = f2bf(q2.z); af[11] = f2bf(q2.w);
      af[12] = f2bf(q3.x); af[13] = f2bf(q3.y); af[14] = f2bf(q3.z); af[15] = f2bf(q3.w);
    }
    #pragma unroll
    for (int j = 0; j < 4; ++j) {
      int n = ntq * 64 + j * 16 + lr;
      const float* wrow = W + (size_t)n * K + kb + khalf * 16;
      __builtin_prefetch(wrow + 32, 0, 1);  // stream next K-chunk (global_prefetch_b8)
      const float4* wp = reinterpret_cast<const float4*>(wrow);
      float4 r0 = wp[0], r1 = wp[1], r2 = wp[2], r3 = wp[3];
      v16bf bfrag;
      bfrag[0]  = f2bf(r0.x); bfrag[1]  = f2bf(r0.y); bfrag[2]  = f2bf(r0.z); bfrag[3]  = f2bf(r0.w);
      bfrag[4]  = f2bf(r1.x); bfrag[5]  = f2bf(r1.y); bfrag[6]  = f2bf(r1.z); bfrag[7]  = f2bf(r1.w);
      bfrag[8]  = f2bf(r2.x); bfrag[9]  = f2bf(r2.y); bfrag[10] = f2bf(r2.z); bfrag[11] = f2bf(r2.w);
      bfrag[12] = f2bf(r3.x); bfrag[13] = f2bf(r3.y); bfrag[14] = f2bf(r3.z); bfrag[15] = f2bf(r3.w);
      acc[j] = __builtin_amdgcn_wmma_f32_16x16x32_bf16(
          false, af, false, bfrag, (short)0, acc[j], false, false);
    }
  }

  #pragma unroll
  for (int j = 0; j < 4; ++j) {
    int n = ntq * 64 + j * 16 + lr;
    float bv = bias[n];
    #pragma unroll
    for (int r = 0; r < 8; ++r) {
      int m = mt * 16 + r + (khalf << 3);
      out[(size_t)m * N + n] = acc[j][r] + bv;
    }
  }
}

// =====================================================================
extern "C" void kernel_launch(void* const* d_in, const int* in_sizes, int n_in,
                              void* d_out, int out_size, void* d_ws, size_t ws_size,
                              hipStream_t stream) {
  const float* x      = (const float*)d_in[0];
  const float* w_in   = (const float*)d_in[1];
  const float* b_in   = (const float*)d_in[2];
  const float* w_cell = (const float*)d_in[3];
  const float* w1 = (const float*)d_in[4];  const float* b1 = (const float*)d_in[5];
  const float* w2 = (const float*)d_in[6];  const float* b2 = (const float*)d_in[7];
  const float* w3 = (const float*)d_in[8];  const float* b3 = (const float*)d_in[9];
  const float* w4 = (const float*)d_in[10]; const float* b4 = (const float*)d_in[11];
  const float* w5 = (const float*)d_in[12]; const float* b5 = (const float*)d_in[13];
  float* out = (float*)d_out;

  uint8_t* ws = (uint8_t*)d_ws;
  size_t off = 0;
  auto alloc = [&](size_t bytes) -> void* {
    void* p = ws + off;
    off = (off + bytes + 255) & ~(size_t)255;
    return p;
  };
  float* cflat = (float*)alloc((size_t)BB * 4096 * sizeof(float));  // c_T, h-major
  float* a1    = (float*)alloc((size_t)BB * 2048 * sizeof(float));
  float* a2    = (float*)alloc((size_t)BB * 1024 * sizeof(float));
  float* a3    = (float*)alloc((size_t)BB * 512 * sizeof(float));
  float* a4    = (float*)alloc((size_t)BB * 256 * sizeof(float));

  // Entire ConvLSTM (input conv + 128-step scan) in a single launch.
  lstm_scan<<<BB, 256, SMEM_TOTAL, stream>>>(cflat, x, w_in, b_in, w_cell);

  mlp_gemm<<<(128 / 16) * (2048 / 64), 32, 0, stream>>>(cflat, w1, b1, a1, 128, 2048, 4096);
  mlp_gemm<<<(128 / 16) * (1024 / 64), 32, 0, stream>>>(a1, w2, b2, a2, 128, 1024, 2048);
  mlp_gemm<<<(128 / 16) * (512 / 64),  32, 0, stream>>>(a2, w3, b3, a3, 128, 512, 1024);
  mlp_gemm<<<(128 / 16) * (256 / 64),  32, 0, stream>>>(a3, w4, b4, a4, 128, 256, 512);
  mlp_gemm<<<(128 / 16) * (128 / 64),  32, 0, stream>>>(a4, w5, b5, out, 128, 128, 256);
}
